// birealnet_binary_conv_55705725829476
// MI455X (gfx1250) — matrix operand order repaired and checked by reading, not compiled, not run on hardware
//
#include <hip/hip_runtime.h>

// ---------------------------------------------------------------------------
// BiRealNet binary conv (3x3, SAME, NHWC 32x56x56x256 -> x256) for gfx1250.
// Forward math:  out = conv(x, sign(w)*mask) * scale[co],  scale = mean|w|.
// Implicit GEMM: M=100352 (=784*128), K=2304 (=9*256), N=256, bf16 WMMA with
// f32 accumulation; weights are exact {-1,0,+1} in bf16, per-filter f32 scale
// applied in the epilogue. Compute-bound (~24us @ ~5PF bf16 dense vs ~9us HBM).
// x (103MB) fits in 192MB L2 -> 9x halo re-reads are L2 hits; output stores are
// non-temporal so they don't evict x. Flat 72-step K loop, register-prefetch
// software pipeline, double-buffered LDS (32KB), hardware bf16 converts.
// ---------------------------------------------------------------------------

typedef __attribute__((ext_vector_type(16))) __bf16 v16bf;
typedef __attribute__((ext_vector_type(4)))  __bf16 v4bf;
typedef __attribute__((ext_vector_type(4)))  float  v4fv;
typedef __attribute__((ext_vector_type(8)))  float  v8f;

#define KVOL 2304
#define NCO  256
#define MTOT 100352   // 32*56*56

__device__ __forceinline__ unsigned short f2bf(float f) {
  unsigned int u = __float_as_uint(f);
  u += 0x7FFFu + ((u >> 16) & 1u);          // round-to-nearest-even
  return (unsigned short)(u >> 16);
}

// ---------------------------------------------------------------------------
// Kernel 1: per-filter scale = mean|w| ; Wt[co][k] = bf16(sign(w)*mask)
// w is (3,3,256,256) row-major => w[k*256+co], k = (kh*3+kw)*256+ci.
// ---------------------------------------------------------------------------
__global__ void birealnet_prep_kernel(const float* __restrict__ w,
                                      const float* __restrict__ mask,
                                      unsigned short* __restrict__ Wt,
                                      float* __restrict__ scale) {
  const int co = blockIdx.x;
  const int t  = threadIdx.x;                 // 256 threads
  __shared__ float red[256];
  float s = 0.f;
  for (int k = t; k < KVOL; k += 256) s += fabsf(w[k * NCO + co]);
  red[t] = s;
  __syncthreads();
  for (int off = 128; off > 0; off >>= 1) {
    if (t < off) red[t] += red[t + off];
    __syncthreads();
  }
  if (t == 0) scale[co] = red[0] * (1.0f / (float)KVOL);
  for (int k = t; k < KVOL; k += 256) {
    float v  = w[k * NCO + co];
    float sg = (v > 0.f) ? 1.f : ((v < 0.f) ? -1.f : 0.f);
    Wt[co * KVOL + k] = f2bf(sg * mask[k * NCO + co]);
  }
}

// ---------------------------------------------------------------------------
// Kernel 2: implicit-GEMM conv with bf16 WMMA, software pipelined.
// Block = 256 threads (8 wave32). Tile: 128 M x 128 N per workgroup.
// Waves: mgrp = wave>>1 (4 M-groups, 2 tiles of 16 rows each),
//        nhalf = wave&1 (64 filters, 4 tiles of 16).
// Flat K loop: s = 0..71, k = 32*s, tap p = s>>3, ci0 = (s&7)*32.
// ---------------------------------------------------------------------------
__global__ __launch_bounds__(256) void birealnet_conv_wmma_kernel(
    const float* __restrict__ x,
    const unsigned short* __restrict__ Wt,
    const float* __restrict__ scale,
    float* __restrict__ out) {
  __shared__ unsigned short As[2][128 * 32];   // 8KB x2: act tile, bf16
  __shared__ unsigned short Bs[2][128 * 32];   // 8KB x2: weight tile, bf16

  const int tid   = threadIdx.x;
  const int lane  = tid & 31;
  const int wave  = tid >> 5;
  const int khalf = lane >> 4;    // 0/1 -> K-half of fragment
  const int l16   = lane & 15;
  const int mgrp  = wave >> 1;    // 0..3
  const int nhalf = wave & 1;     // 0..1

  const int Mbase = blockIdx.x * 128;          // 784 blocks: exact
  const int Nbase = blockIdx.y * 128;          // 2 blocks: exact

  // --- loader roles (all 256 threads) ---
  // A: thread covers spatial row lrow (0..127), 16 ci values per step.
  const int lrow = tid >> 1;
  const int lcg  = tid & 1;
  const int M    = Mbase + lrow;
  const int nImg = M / 3136;
  const int rem  = M - nImg * 3136;
  const int h    = rem / 56;
  const int wc   = rem - h * 56;
  const float* xptr = x + (nImg * 3136 + h * 56 + wc) * NCO + lcg * 16;
  // 9-bit SAME-padding validity mask, one bit per tap p = (dh+1)*3 + (dw+1).
  unsigned vmask = 0;
#pragma unroll
  for (int p = 0; p < 9; ++p) {
    const int hh = h + p / 3 - 1;
    const int ww = wc + p % 3 - 1;
    if (((unsigned)hh < 56u) && ((unsigned)ww < 56u)) vmask |= (1u << p);
  }
  // B: thread covers filter row bco; k walks the Wt row linearly (k = 32*s).
  const int bco = tid >> 1;
  const unsigned short* wtRow = Wt + (Nbase + bco) * KVOL + lcg * 16;

  v8f acc[2][4];
  const v8f vzero = {0.f, 0.f, 0.f, 0.f, 0.f, 0.f, 0.f, 0.f};
#pragma unroll
  for (int mi = 0; mi < 2; ++mi)
#pragma unroll
    for (int j = 0; j < 4; ++j) acc[mi][j] = vzero;

  // ---- register staging for the software pipeline ----
  uint4 ra0, ra1, rb0, rb1;
  const uint4 z4 = make_uint4(0u, 0u, 0u, 0u);

  auto preload = [&](int s) {
    const int p   = s >> 3;                       // uniform (SALU)
    const int ci0 = (s & 7) << 5;                 // uniform (SALU)
    // B: pure linear copy of the Wt row.
    const unsigned short* bsrc = wtRow + s * 32;
    rb0 = ((const uint4*)bsrc)[0];
    rb1 = ((const uint4*)bsrc)[1];
    // A: tap-shifted x row, hardware bf16 convert.
    if ((vmask >> p) & 1u) {
      const int delta = ((p / 3 - 1) * 56 + (p % 3 - 1)) * NCO;  // uniform
      const float* sp = xptr + delta + ci0;
      v4fv f0 = ((const v4fv*)sp)[0];
      v4fv f1 = ((const v4fv*)sp)[1];
      v4fv f2 = ((const v4fv*)sp)[2];
      v4fv f3 = ((const v4fv*)sp)[3];
      union { v4bf h[2]; uint4 u; } c0, c1;
      c0.h[0] = __builtin_convertvector(f0, v4bf);
      c0.h[1] = __builtin_convertvector(f1, v4bf);
      c1.h[0] = __builtin_convertvector(f2, v4bf);
      c1.h[1] = __builtin_convertvector(f3, v4bf);
      ra0 = c0.u;
      ra1 = c1.u;
    } else {
      ra0 = z4;
      ra1 = z4;
    }
  };

  preload(0);
  for (int s = 0; s < 72; ++s) {
    const int buf = s & 1;
    // ---- commit staged registers to LDS ----
    {
      unsigned short* adst = &As[buf][lrow * 32 + lcg * 16];
      ((uint4*)adst)[0] = ra0;
      ((uint4*)adst)[1] = ra1;
      unsigned short* bdst = &Bs[buf][bco * 32 + lcg * 16];
      ((uint4*)bdst)[0] = rb0;
      ((uint4*)bdst)[1] = rb1;
    }
    __syncthreads();
    // ---- issue next step's global loads (latency overlaps the WMMAs) ----
    if (s < 71) preload(s + 1);
    // ---- A fragments: 16x32 bf16, per-lane K {0-7,16-23}/{8-15,24-31} ----
    union { v16bf v; uint4 u[2]; } af[2];
#pragma unroll
    for (int mi = 0; mi < 2; ++mi) {
      const unsigned short* ap = &As[buf][(mgrp * 32 + mi * 16 + l16) * 32];
      af[mi].u[0] = *(const uint4*)(ap + khalf * 8);
      af[mi].u[1] = *(const uint4*)(ap + 16 + khalf * 8);
    }
    // ---- B fragments + 8 WMMAs ----
#pragma unroll
    for (int j = 0; j < 4; ++j) {
      const unsigned short* bp =
          &Bs[buf][(nhalf * 64 + j * 16 + l16) * 32 + khalf * 8];
      union { v16bf v; uint4 u[2]; } bf_;
      bf_.u[0] = ((const uint4*)bp)[0];
      bf_.u[1] = ((const uint4*)bp)[2];
#pragma unroll
      for (int mi = 0; mi < 2; ++mi) {
        acc[mi][j] = __builtin_amdgcn_wmma_f32_16x16x32_bf16(
            false, af[mi].v, false, bf_.v, (short)0, acc[mi][j],
            false, false);
      }
    }
  }

  // ---- epilogue: apply per-filter scale, non-temporal f32 stores ----
#pragma unroll
  for (int j = 0; j < 4; ++j) {
    const int co = Nbase + nhalf * 64 + j * 16 + l16;
    const float sc = scale[co];
#pragma unroll
    for (int mi = 0; mi < 2; ++mi) {
      const int m0 = Mbase + mgrp * 32 + mi * 16 + khalf * 8;
#pragma unroll
      for (int v = 0; v < 8; ++v) {
        __builtin_nontemporal_store(acc[mi][j][v] * sc,
                                    &out[(m0 + v) * NCO + co]);
      }
    }
  }
}

// ---------------------------------------------------------------------------
extern "C" void kernel_launch(void* const* d_in, const int* in_sizes, int n_in,
                              void* d_out, int out_size, void* d_ws, size_t ws_size,
                              hipStream_t stream) {
  const float* x    = (const float*)d_in[0];   // (32,56,56,256) f32
  const float* w    = (const float*)d_in[1];   // (3,3,256,256)  f32
  const float* mask = (const float*)d_in[2];   // (2304,256)     f32
  float* out        = (float*)d_out;           // (32,56,56,256) f32

  // workspace: Wt bf16 [256][2304] (1.18MB) then scale f32 [256]
  unsigned short* Wt = (unsigned short*)d_ws;
  float* scale = (float*)((char*)d_ws + (size_t)NCO * KVOL * sizeof(unsigned short));

  birealnet_prep_kernel<<<dim3(NCO), dim3(256), 0, stream>>>(w, mask, Wt, scale);

  dim3 grid(MTOT / 128, NCO / 128);            // 784 x 2
  birealnet_conv_wmma_kernel<<<grid, dim3(256), 0, stream>>>(x, Wt, scale, out);
}
